// ScoreFeatureExtractor_317827580764
// MI455X (gfx1250) — compile-verified, matrix operand order checked by reading, not run on hardware
//
#include <hip/hip_runtime.h>
#include <cstdint>
#include <cstddef>

// ---------------------------------------------------------------------------
// CDNA5 (gfx1250) implementation: bf16 WMMA GEMMs for all dense math,
// persistent-workgroup LSTM recurrence with h/gates resident in LDS (c in registers) and a
// TDM (tensor_load_to_lds) - cached slice of Whh exploiting the 320KB WGP LDS.
// Workspace requirement: ~0.6 GB (intermediates reuse regions where legal).
// ---------------------------------------------------------------------------

typedef __attribute__((ext_vector_type(16))) __bf16 bf16x16;
typedef __attribute__((ext_vector_type(8)))  __bf16 bf16x8;
typedef __attribute__((ext_vector_type(8)))  float  f32x8;
typedef __attribute__((ext_vector_type(4)))  unsigned int u32x4;
typedef __attribute__((ext_vector_type(4)))  int i32x4;
typedef __attribute__((ext_vector_type(8)))  int i32x8;

#define WMMA_BF16(a, b, c) \
  __builtin_amdgcn_wmma_f32_16x16x32_bf16(false, (a), false, (b), (short)0, (c), false, false)

__device__ __forceinline__ float sigf(float x) { return 1.0f / (1.0f + __expf(-x)); }

// Problem constants
constexpr int kB  = 32;     // batch
constexpr int kL  = 4096;   // phoneme length
constexpr int kN  = 2048;   // notes
constexpr int kD  = 256;    // channels / hidden
constexpr int kG4 = 1024;   // 4*hidden (gates)

// ---------------------------------------------------------------------------
// Weight packing kernels (f32 -> bf16, padded-K layouts)
// ---------------------------------------------------------------------------

// src [CoutS][Cin][3] -> dst [3][CoutD][Kp]  (zero pad Cin->Kp, CoutS->CoutD)
__global__ void k_pack_conv(const float* __restrict__ src, __bf16* __restrict__ dst,
                            int CoutS, int CoutD, int Cin, int Kp) {
  int total = 3 * CoutD * Kp;
  for (int i = blockIdx.x * blockDim.x + threadIdx.x; i < total; i += gridDim.x * blockDim.x) {
    int s   = i / (CoutD * Kp);
    int rem = i - s * CoutD * Kp;
    int co  = rem / Kp;
    int ci  = rem - co * Kp;
    float v = (co < CoutS && ci < Cin) ? src[((size_t)co * Cin + ci) * 3 + s] : 0.f;
    dst[i] = (__bf16)v;
  }
}

// src [R][C] -> dst [R][Cp] bf16 (zero pad C->Cp)
__global__ void k_pack_mat(const float* __restrict__ src, __bf16* __restrict__ dst,
                           int R, int C, int Cp) {
  int total = R * Cp;
  for (int i = blockIdx.x * blockDim.x + threadIdx.x; i < total; i += gridDim.x * blockDim.x) {
    int r = i / Cp, c = i - r * Cp;
    dst[i] = (__bf16)((c < C) ? src[(size_t)r * C + c] : 0.f);
  }
}

__global__ void k_bias_sum(const float* __restrict__ a, const float* __restrict__ b,
                           float* __restrict__ o, int n) {
  for (int i = blockIdx.x * blockDim.x + threadIdx.x; i < n; i += gridDim.x * blockDim.x)
    o[i] = a[i] + b[i];
}

// ---------------------------------------------------------------------------
// Embedding gather: dst [B][L][288] bf16  (cols 0..255 emb, 256..265 pos, rest 0)
// ---------------------------------------------------------------------------
__global__ void k_embed(const int* __restrict__ seq, const int* __restrict__ order,
                        const float* __restrict__ embW, const float* __restrict__ posW,
                        __bf16* __restrict__ dst) {
  const size_t total = (size_t)kB * kL * 288;
  const size_t stride = (size_t)gridDim.x * blockDim.x;
  for (size_t i = (size_t)blockIdx.x * blockDim.x + threadIdx.x; i < total; i += stride) {
    int c = (int)(i % 288);
    size_t bl = i / 288;
    float v;
    if (c < 256)      v = embW[(size_t)seq[bl] * kD + c];
    else if (c < 266) v = posW[(size_t)order[bl] * 10 + (c - 256)];
    else              v = 0.f;
    dst[i] = (__bf16)v;
  }
}

// ---------------------------------------------------------------------------
// Run (note segment) detection: serial scan per batch row (4096 steps, cheap)
// ---------------------------------------------------------------------------
__global__ void k_runs(const int* __restrict__ seq, int* __restrict__ start,
                       int* __restrict__ len, int* __restrict__ nruns) {
  int b = blockIdx.x;
  if (threadIdx.x != 0) return;
  int n = -1;
  bool prev = false;
  for (int l = 0; l < kL; ++l) {
    bool m = seq[(size_t)b * kL + l] > 1;
    if (m && !prev) {
      ++n;
      if (n < kN) { start[b * kN + n] = l; len[b * kN + n] = 0; }
    }
    if (m && n < kN) len[b * kN + n]++;
    prev = m;
  }
  int r = n + 1;
  nruns[b] = (r < kN) ? r : kN;
}

// ---------------------------------------------------------------------------
// Segment mean + feature concat: xl0 [B][N][288] bf16
//   cols 0..255 mean(x2 over run), 256 = dur, 257 = 1/(dur+1), 258..287 = 0
// ---------------------------------------------------------------------------
__global__ void k_agg(const __bf16* __restrict__ x2, const float* __restrict__ dur,
                      const int* __restrict__ start, const int* __restrict__ len,
                      const int* __restrict__ nruns, __bf16* __restrict__ xl0) {
  const int n = blockIdx.x, b = blockIdx.y, d = threadIdx.x;  // 256 threads
  float v = 0.f;
  if (n < nruns[b]) {
    int s = start[b * kN + n], e = len[b * kN + n];
    float acc = 0.f;
    for (int l = 0; l < e; ++l)
      acc += (float)x2[(((size_t)b * kL) + (s + l)) * kD + d];
    v = acc / (float)e;
  }
  size_t o = (((size_t)b * kN) + n) * 288;
  xl0[o + d] = (__bf16)v;
  if (d < 32) {
    float sd = dur[(size_t)b * kN + n];
    float e2 = (d == 0) ? sd : (d == 1) ? 1.f / (sd + 1.f) : 0.f;
    xl0[o + 256 + d] = (__bf16)e2;
  }
}

// ---------------------------------------------------------------------------
// Generic bf16 WMMA GEMM.   C[m][n] (+)= sum_s sum_k A[m+s-pad][k] * B_s[n][k]
// A: bf16 row-major [*, lda]; B: bf16 "torch weight" layout [S][NB][ldb].
// Epilogue: + bias[n], optional relu, store f32 or bf16, only n < Nstore.
// Block: 256 threads = 8 wave32 waves (2x4), C-tile 64(M) x 128(N).
// ---------------------------------------------------------------------------
struct GemmP {
  const __bf16* A;  size_t aBatch; int lda; int MA;
  const __bf16* Bm; size_t bShift; int ldb; int NB;
  float* Cf; __bf16* Cb; size_t cBatch; int ldc;
  const float* bias;
  int M; int Nstore; int Ktiles; int S; int relu;
};

__global__ __launch_bounds__(256) void k_gemm(GemmP p) {
  __shared__ __bf16 As[64 * 48];   // 64 rows x 32 k, stride 48 (32B-aligned rows)
  __shared__ __bf16 Bs[128 * 48];

  const int tid = threadIdx.x;
  const int lane = tid & 31, wave = tid >> 5;
  const int wm = wave >> 2, wn = wave & 3;          // 2 x 4 wave grid
  const int b = blockIdx.z;
  const int m0 = blockIdx.y * 64, n0 = blockIdx.x * 128;
  const __bf16* A = p.A + (size_t)b * p.aBatch;
  const int pad = (p.S == 3) ? 1 : 0;

  f32x8 zero = {};
  f32x8 acc[2][2];
  acc[0][0] = zero; acc[0][1] = zero; acc[1][0] = zero; acc[1][1] = zero;

  const int ar = tid >> 2, akc = (tid & 3) * 8;      // A stage: 64 rows x 4 chunks
  const int br = tid >> 1, bkc = (tid & 1) * 16;     // B stage: 128 rows x 2 chunks
  const int lrow = lane & 15, khalf = (lane >> 4) * 16, mhalf = (lane >> 4) * 8;

  for (int s = 0; s < p.S; ++s) {
    const __bf16* Bp = p.Bm + (size_t)s * p.bShift;
    for (int kt = 0; kt < p.Ktiles; ++kt) {
      {  // stage A tile (64 x 32), with conv shift + bounds masking
        int gr = m0 + ar + s - pad;
        bf16x8 v = {};
        if (gr >= 0 && gr < p.MA) {
          v = *(const bf16x8*)&A[(size_t)gr * p.lda + kt * 32 + akc];
          if (kt + 1 < p.Ktiles)  // L2 prefetch of next k-tile (global_prefetch_b8)
            __builtin_prefetch(&A[(size_t)gr * p.lda + (kt + 1) * 32 + akc], 0, 1);
        }
        *(bf16x8*)&As[ar * 48 + akc] = v;
      }
      {  // stage B tile (128 x 32), zero-fill rows >= NB
        int gn = n0 + br;
        bf16x8 v0 = {}, v1 = {};
        if (gn < p.NB) {
          const __bf16* src = &Bp[(size_t)gn * p.ldb + kt * 32 + bkc];
          v0 = *(const bf16x8*)src;
          v1 = *(const bf16x8*)(src + 8);
        }
        *(bf16x8*)&Bs[br * 48 + bkc]     = v0;
        *(bf16x8*)&Bs[br * 48 + bkc + 8] = v1;
      }
      __syncthreads();
      bf16x16 a0 = *(const bf16x16*)&As[(wm * 32 + lrow) * 48 + khalf];
      bf16x16 a1 = *(const bf16x16*)&As[(wm * 32 + 16 + lrow) * 48 + khalf];
      bf16x16 b0 = *(const bf16x16*)&Bs[(wn * 32 + lrow) * 48 + khalf];
      bf16x16 b1 = *(const bf16x16*)&Bs[(wn * 32 + 16 + lrow) * 48 + khalf];
      acc[0][0] = WMMA_BF16(a0, b0, acc[0][0]);
      acc[0][1] = WMMA_BF16(a0, b1, acc[0][1]);
      acc[1][0] = WMMA_BF16(a1, b0, acc[1][0]);
      acc[1][1] = WMMA_BF16(a1, b1, acc[1][1]);
      __syncthreads();
    }
  }

  // epilogue (C layout: VGPR r -> M = r + 8*(lane>>4), N = lane&15)
  for (int mi = 0; mi < 2; ++mi)
    for (int ni = 0; ni < 2; ++ni)
      for (int r = 0; r < 8; ++r) {
        int m = m0 + wm * 32 + mi * 16 + mhalf + r;
        int n = n0 + wn * 32 + ni * 16 + (lane & 15);
        if (m < p.M && n < p.Nstore) {
          float v = acc[mi][ni][r];
          if (p.bias) v += p.bias[n];
          if (p.relu) v = fmaxf(v, 0.f);
          size_t o = (size_t)b * p.cBatch + (size_t)m * p.ldc + n;
          if (p.Cf) p.Cf[o] = v; else p.Cb[o] = (__bf16)v;
        }
      }
}

// ---------------------------------------------------------------------------
// LSTM recurrence: one persistent 32-wave workgroup per direction.
//   gates[32 x 1024] = pre[t] + h[32 x 256] @ Whh^T   (WMMA bf16)
//   h (17KB) + gates (132KB) + TDM-cached Whh k-tiles 0..1 (160KB) all live in
//   the 320KB WGP LDS; c stays in registers. Remaining 6 Whh k-tiles (384KB)
//   stream from the 192MB L2 each step.
// grid.x = 2: block 0 forward, block 1 backward (independent, concurrent).
// ---------------------------------------------------------------------------
__global__ __launch_bounds__(1024) void k_lstm(
    const __bf16* __restrict__ pre_f, const __bf16* __restrict__ pre_b,
    const __bf16* __restrict__ whh_f, const __bf16* __restrict__ whh_b,
    __bf16* __restrict__ hs, int T) {
  __shared__ __bf16 hsh[32 * 272];     // h (bf16), padded stride
  __shared__ float  gsh[32 * 1032];    // gates (f32), padded stride
  __shared__ __bf16 wsh[1024 * 80];    // Whh[:, 0:64] cached via TDM, 80-elem rows

  const int tid = threadIdx.x, lane = tid & 31, wave = tid >> 5;
  const int rev = blockIdx.x;
  const __bf16* pre = rev ? pre_b : pre_f;
  const __bf16* whh = rev ? whh_b : whh_f;
  const int coloff = rev ? 256 : 0;
  const int n0 = wave * 32;                                   // this wave's gate cols
  const int lrow = lane & 15, khalf = (lane >> 4) * 16, mhalf = (lane >> 4) * 8;
  const int eb = tid >> 5, eu = (tid & 31) * 8;               // elementwise ownership

  // --- TDM: DMA Whh rows 0..1023, cols 0..63 into LDS (one issue, wave 0) ---
  // D# group0: count=1, lds_addr, global_addr[56:0], type=2 ("image").
  // D# group1: data_size=2B, pad_enable, pad_interval=32 dwords (code 4),
  //            pad_amount=8 dwords (code 7) -> LDS row stride 80 bf16,
  //            tensor_dim0=256, tensor_dim1=1024, tile=[64 x 1024], stride0=256.
  if (wave == 0) {
    unsigned long long ga = (unsigned long long)(uintptr_t)whh;
    unsigned int lds = (unsigned int)(uintptr_t)wsh;
    u32x4 g0;
    g0[0] = 1u;                                            // count=1 (user D#)
    g0[1] = lds;                                           // lds_addr
    g0[2] = (unsigned int)ga;                              // global_addr[31:0]
    g0[3] = (unsigned int)((ga >> 32) & 0x01FFFFFFu) | (2u << 30);  // [56:32] | type=2
    i32x8 g1;
    g1[0] = (int)((1u << 16) | (1u << 20) | (4u << 22) | (7u << 25));
    g1[1] = (int)(256u << 16);   // tensor_dim0 = 256  (bits 79:48, low half)
    g1[2] = (int)(1024u << 16);  // tensor_dim1 = 1024 (bits 111:80, low half)
    g1[3] = (int)(64u << 16);    // tile_dim0 = 64     (bits 127:112)
    g1[4] = 1024;                // tile_dim1 = 1024   (bits 143:128)
    g1[5] = 256;                 // tensor_dim0_stride = 256 (bits 207:160, low 32)
    g1[6] = 0;
    g1[7] = 0;
    i32x4 gz = {0, 0, 0, 0};
#if defined(__clang_major__) && (__clang_major__ >= 23)
    i32x8 gz8 = {0, 0, 0, 0, 0, 0, 0, 0};
    __builtin_amdgcn_tensor_load_to_lds(g0, g1, gz, gz, gz8, 0);
#else
    __builtin_amdgcn_tensor_load_to_lds(g0, g1, gz, gz, 0);
#endif
    __builtin_amdgcn_s_wait_tensorcnt(0);
  }

  for (int i = tid; i < 32 * 272; i += 1024) hsh[i] = (__bf16)0.f;
  float c[8];
#pragma unroll
  for (int q = 0; q < 8; ++q) c[q] = 0.f;
  __syncthreads();   // h zeroed AND TDM weight cache visible to all waves

  for (int ts = 0; ts < T; ++ts) {
    const int t = rev ? (T - 1 - ts) : ts;

    // init accumulators from precomputed input projection (already biased)
    f32x8 acc[2][2];
#pragma unroll
    for (int mi = 0; mi < 2; ++mi)
#pragma unroll
      for (int ni = 0; ni < 2; ++ni) {
        f32x8 a;
#pragma unroll
        for (int r = 0; r < 8; ++r)
          a[r] = (float)pre[((size_t)(mi * 16 + mhalf + r) * T + t) * kG4 +
                            (n0 + ni * 16 + (lane & 15))];
        acc[mi][ni] = a;
      }
    if (ts + 1 < T) {  // L2 prefetch of next timestep's pre slice
      const int tn = rev ? (T - 2 - ts) : (ts + 1);
      __builtin_prefetch(&pre[((size_t)mhalf * T + tn) * kG4 + n0 + (lane & 15)], 0, 1);
    }

    // gates += h @ Whh^T   (K = 256; k-tiles 0..1 from LDS cache, rest from L2)
#pragma unroll
    for (int kt = 0; kt < 8; ++kt) {
      const int k = kt * 32 + khalf;
      bf16x16 a0 = *(const bf16x16*)&hsh[lrow * 272 + k];
      bf16x16 a1 = *(const bf16x16*)&hsh[(16 + lrow) * 272 + k];
      bf16x16 b0, b1;
      if (kt < 2) {
        b0 = *(const bf16x16*)&wsh[(n0 + lrow) * 80 + k];
        b1 = *(const bf16x16*)&wsh[(n0 + 16 + lrow) * 80 + k];
      } else {
        b0 = *(const bf16x16*)&whh[(size_t)(n0 + lrow) * kD + k];
        b1 = *(const bf16x16*)&whh[(size_t)(n0 + 16 + lrow) * kD + k];
      }
      acc[0][0] = WMMA_BF16(a0, b0, acc[0][0]);
      acc[0][1] = WMMA_BF16(a0, b1, acc[0][1]);
      acc[1][0] = WMMA_BF16(a1, b0, acc[1][0]);
      acc[1][1] = WMMA_BF16(a1, b1, acc[1][1]);
    }

    // spill gates to LDS for the cross-gate elementwise update
#pragma unroll
    for (int mi = 0; mi < 2; ++mi)
#pragma unroll
      for (int ni = 0; ni < 2; ++ni)
#pragma unroll
        for (int r = 0; r < 8; ++r)
          gsh[(mi * 16 + mhalf + r) * 1032 + n0 + ni * 16 + (lane & 15)] = acc[mi][ni][r];
    __syncthreads();

    // c' = sig(f)*c + sig(i)*tanh(g);  h = sig(o)*tanh(c')
#pragma unroll
    for (int q = 0; q < 8; ++q) {
      const int u = eu + q;
      float ig = gsh[eb * 1032 + u];
      float fg = gsh[eb * 1032 + 256 + u];
      float gg = gsh[eb * 1032 + 512 + u];
      float og = gsh[eb * 1032 + 768 + u];
      float cn = sigf(fg) * c[q] + sigf(ig) * tanhf(gg);
      c[q] = cn;
      float hv = sigf(og) * tanhf(cn);
      hsh[eb * 272 + u] = (__bf16)hv;
      hs[((size_t)eb * T + t) * 512 + coloff + u] = (__bf16)hv;
    }
    __syncthreads();
  }
}

// ---------------------------------------------------------------------------
// Host orchestration
// ---------------------------------------------------------------------------
static inline void launch_gemm(hipStream_t s, const GemmP& p, int Npad) {
  dim3 g((unsigned)((Npad + 127) / 128), (unsigned)((p.M + 63) / 64), kB);
  k_gemm<<<g, 256, 0, s>>>(p);
}

extern "C" void kernel_launch(void* const* d_in, const int* in_sizes, int n_in,
                              void* d_out, int out_size, void* d_ws, size_t ws_size,
                              hipStream_t stream) {
  (void)in_sizes; (void)n_in; (void)out_size; (void)ws_size;

  const float* score = (const float*)d_in[0];
  const int*   seq   = (const int*)d_in[1];
  const int*   order = (const int*)d_in[2];
  const float* embW  = (const float*)d_in[3];
  const float* posW  = (const float*)d_in[4];
  const float* mixW1 = (const float*)d_in[5];
  const float* mixb1 = (const float*)d_in[6];
  const float* mixW2 = (const float*)d_in[7];
  const float* mixb2 = (const float*)d_in[8];
  const float* wih0f = (const float*)d_in[9];
  const float* whh0f = (const float*)d_in[10];
  const float* bih0f = (const float*)d_in[11];
  const float* bhh0f = (const float*)d_in[12];
  const float* wih0b = (const float*)d_in[13];
  const float* whh0b = (const float*)d_in[14];
  const float* bih0b = (const float*)d_in[15];
  const float* bhh0b = (const float*)d_in[16];
  const float* wih1f = (const float*)d_in[17];
  const float* whh1f = (const float*)d_in[18];
  const float* bih1f = (const float*)d_in[19];
  const float* bhh1f = (const float*)d_in[20];
  const float* wih1b = (const float*)d_in[21];
  const float* whh1b = (const float*)d_in[22];
  const float* bih1b = (const float*)d_in[23];
  const float* bhh1b = (const float*)d_in[24];
  const float* cW1   = (const float*)d_in[25];
  const float* cb1   = (const float*)d_in[26];
  const float* cW2   = (const float*)d_in[27];
  const float* cb2   = (const float*)d_in[28];

  // -------- workspace layout (deterministic every call) --------
  size_t off = 0;
  auto alloc = [&](size_t bytes) -> char* {
    char* p = (char*)d_ws + off;
    off += (bytes + 255) & ~(size_t)255;
    return p;
  };
  __bf16* w_mix1 = (__bf16*)alloc((size_t)3 * 256 * 288 * 2);
  __bf16* w_mix2 = (__bf16*)alloc((size_t)3 * 256 * 256 * 2);
  __bf16* w_c1   = (__bf16*)alloc((size_t)3 * 256 * 512 * 2);
  __bf16* w_c2   = (__bf16*)alloc((size_t)3 * 16 * 256 * 2);
  __bf16* w_ih0f = (__bf16*)alloc((size_t)kG4 * 288 * 2);
  __bf16* w_ih0b = (__bf16*)alloc((size_t)kG4 * 288 * 2);
  __bf16* w_ih1f = (__bf16*)alloc((size_t)kG4 * 512 * 2);
  __bf16* w_ih1b = (__bf16*)alloc((size_t)kG4 * 512 * 2);
  __bf16* w_hh0f = (__bf16*)alloc((size_t)kG4 * kD * 2);
  __bf16* w_hh0b = (__bf16*)alloc((size_t)kG4 * kD * 2);
  __bf16* w_hh1f = (__bf16*)alloc((size_t)kG4 * kD * 2);
  __bf16* w_hh1b = (__bf16*)alloc((size_t)kG4 * kD * 2);
  float* bias0f = (float*)alloc(kG4 * 4);
  float* bias0b = (float*)alloc(kG4 * 4);
  float* bias1f = (float*)alloc(kG4 * 4);
  float* bias1b = (float*)alloc(kG4 * 4);

  __bf16* emb  = (__bf16*)alloc((size_t)kB * kL * 288 * 2);   // reused later for hs1
  __bf16* act1 = (__bf16*)alloc((size_t)kB * kL * kD * 2);    // reused later for c1a
  __bf16* x2   = (__bf16*)alloc((size_t)kB * kL * kD * 2);
  int* seg_start = (int*)alloc((size_t)kB * kN * 4);
  int* seg_len   = (int*)alloc((size_t)kB * kN * 4);
  int* seg_nruns = (int*)alloc((size_t)kB * 4);
  __bf16* xl0  = (__bf16*)alloc((size_t)kB * kN * 288 * 2);
  __bf16* pre_a = (__bf16*)alloc((size_t)kB * kN * kG4 * 2);  // pre l0f, then l1f
  __bf16* pre_b = (__bf16*)alloc((size_t)kB * kN * kG4 * 2);  // pre l0b, then l1b
  __bf16* hs0  = (__bf16*)alloc((size_t)kB * kN * 512 * 2);
  __bf16* hs1  = emb;    // alias: emb dead after mix1 GEMM
  __bf16* c1a  = act1;   // alias: act1 dead after mix2 GEMM

  // -------- weight prep --------
  k_pack_conv<<<256, 256, 0, stream>>>(mixW1, w_mix1, 256, 256, 266, 288);
  k_pack_conv<<<256, 256, 0, stream>>>(mixW2, w_mix2, 256, 256, 256, 256);
  k_pack_conv<<<256, 256, 0, stream>>>(cW1,   w_c1,   256, 256, 512, 512);
  k_pack_conv<<<64,  256, 0, stream>>>(cW2,   w_c2,   1,   16,  256, 256);
  k_pack_mat<<<256, 256, 0, stream>>>(wih0f, w_ih0f, kG4, 258, 288);
  k_pack_mat<<<256, 256, 0, stream>>>(wih0b, w_ih0b, kG4, 258, 288);
  k_pack_mat<<<256, 256, 0, stream>>>(wih1f, w_ih1f, kG4, 512, 512);
  k_pack_mat<<<256, 256, 0, stream>>>(wih1b, w_ih1b, kG4, 512, 512);
  k_pack_mat<<<256, 256, 0, stream>>>(whh0f, w_hh0f, kG4, kD, kD);
  k_pack_mat<<<256, 256, 0, stream>>>(whh0b, w_hh0b, kG4, kD, kD);
  k_pack_mat<<<256, 256, 0, stream>>>(whh1f, w_hh1f, kG4, kD, kD);
  k_pack_mat<<<256, 256, 0, stream>>>(whh1b, w_hh1b, kG4, kD, kD);
  k_bias_sum<<<4, 256, 0, stream>>>(bih0f, bhh0f, bias0f, kG4);
  k_bias_sum<<<4, 256, 0, stream>>>(bih0b, bhh0b, bias0b, kG4);
  k_bias_sum<<<4, 256, 0, stream>>>(bih1f, bhh1f, bias1f, kG4);
  k_bias_sum<<<4, 256, 0, stream>>>(bih1b, bhh1b, bias1b, kG4);

  // -------- embedding + mix convs (as 3-shift GEMMs) --------
  k_embed<<<8192, 256, 0, stream>>>(seq, order, embW, posW, emb);

  GemmP p{};
  // mix conv1: act1 = relu(emb * W1 + b1)   [B,L,256] bf16
  p.A = emb;  p.aBatch = (size_t)kL * 288; p.lda = 288; p.MA = kL;
  p.Bm = w_mix1; p.bShift = (size_t)256 * 288; p.ldb = 288; p.NB = 256;
  p.Cf = nullptr; p.Cb = act1; p.cBatch = (size_t)kL * kD; p.ldc = kD;
  p.bias = mixb1; p.M = kL; p.Nstore = kD; p.Ktiles = 9; p.S = 3; p.relu = 1;
  launch_gemm(stream, p, 256);
  // mix conv2: x2 = act1 * W2 + b2
  p.A = act1; p.lda = 256; p.Bm = w_mix2; p.bShift = (size_t)256 * 256; p.ldb = 256;
  p.Cb = x2; p.bias = mixb2; p.Ktiles = 8; p.relu = 0;
  launch_gemm(stream, p, 256);

  // -------- segment mean -> LSTM layer-0 input --------
  k_runs<<<kB, 32, 0, stream>>>(seq, seg_start, seg_len, seg_nruns);
  k_agg<<<dim3(kN, kB), 256, 0, stream>>>(x2, score, seg_start, seg_len, seg_nruns, xl0);

  // -------- LSTM layer 0: input projections then recurrence --------
  p = GemmP{};
  p.A = xl0; p.aBatch = (size_t)kN * 288; p.lda = 288; p.MA = kN;
  p.Bm = w_ih0f; p.bShift = 0; p.ldb = 288; p.NB = kG4;
  p.Cf = nullptr; p.Cb = pre_a; p.cBatch = (size_t)kN * kG4; p.ldc = kG4;
  p.bias = bias0f; p.M = kN; p.Nstore = kG4; p.Ktiles = 9; p.S = 1; p.relu = 0;
  launch_gemm(stream, p, kG4);
  p.Bm = w_ih0b; p.Cb = pre_b; p.bias = bias0b;
  launch_gemm(stream, p, kG4);
  k_lstm<<<2, 1024, 0, stream>>>(pre_a, pre_b, w_hh0f, w_hh0b, hs0, kN);

  // -------- LSTM layer 1 --------
  p.A = hs0; p.aBatch = (size_t)kN * 512; p.lda = 512;
  p.Bm = w_ih1f; p.ldb = 512; p.Cb = pre_a; p.bias = bias1f; p.Ktiles = 16;
  launch_gemm(stream, p, kG4);
  p.Bm = w_ih1b; p.Cb = pre_b; p.bias = bias1b;
  launch_gemm(stream, p, kG4);
  k_lstm<<<2, 1024, 0, stream>>>(pre_a, pre_b, w_hh1f, w_hh1b, hs1, kN);

  // -------- output convs --------
  p = GemmP{};
  p.A = hs1; p.aBatch = (size_t)kN * 512; p.lda = 512; p.MA = kN;
  p.Bm = w_c1; p.bShift = (size_t)256 * 512; p.ldb = 512; p.NB = 256;
  p.Cf = nullptr; p.Cb = c1a; p.cBatch = (size_t)kN * kD; p.ldc = kD;
  p.bias = cb1; p.M = kN; p.Nstore = kD; p.Ktiles = 16; p.S = 3; p.relu = 1;
  launch_gemm(stream, p, 256);

  p.A = c1a; p.lda = 256;
  p.Bm = w_c2; p.bShift = (size_t)16 * 256; p.ldb = 256; p.NB = 16;
  p.Cf = (float*)d_out; p.Cb = nullptr; p.cBatch = (size_t)kN; p.ldc = 1;
  p.bias = cb2; p.Nstore = 1; p.Ktiles = 8; p.relu = 0;
  launch_gemm(stream, p, 16);
}
